// Attention_65635690217926
// MI455X (gfx1250) — matrix operand order repaired and checked by reading, not compiled, not run on hardware
//
#include <hip/hip_runtime.h>
#include <hip/hip_bf16.h>

// ---------------------------------------------------------------------------
// Problem constants (from reference): B=32, N=1280 (256 template + 1024
// search), C=256, H=8 heads, hd=32, SR=2 -> M=320 kv tokens (64 + 256).
// ---------------------------------------------------------------------------
#define B_    32
#define N_    1280
#define C_    256
#define H_    8
#define HD_   32
#define M_    320
#define NTQ_  256
#define NTKV_ 64
#define LN_EPS 1e-5f
#define QSCALE 0.17677669529663687f   // 1/sqrt(32)

typedef _Float16 v8h  __attribute__((ext_vector_type(8)));
typedef _Float16 v16h __attribute__((ext_vector_type(16)));
typedef float    v8f  __attribute__((ext_vector_type(8)));

// ---------------------------------------------------------------------------
// Fragment loaders matching CDNA5 WMMA VGPR layouts (CDNA5 ISA):
//  A (16x32 f16): lane L<16 holds row L, K in {k0..k0+7} U {k0+16..k0+23};
//                 lane L>=16 holds row L-16, K in {k0+8..15} U {k0+24..31}.
//    -> two contiguous 16B loads per lane at (base + hl*8) and (+16 more).
//  B (32x16 f16): lane L<16 holds col L, K in {k0..k0+7} U {k0+16..k0+23};
//                 lane L>=16 holds col L-16, K in {k0+8..15} U {k0+24..31}.
//    -> same two 16B loads as A, at (base + hl*8) and (+16 more).
// ---------------------------------------------------------------------------
__device__ __forceinline__ v16h load_fragA(const _Float16* p) {
  v8h lo = *(const v8h*)(p);
  v8h hi = *(const v8h*)(p + 16);
  return __builtin_shufflevector(lo, hi, 0,1,2,3,4,5,6,7,8,9,10,11,12,13,14,15);
}
__device__ __forceinline__ v16h load_fragB(const _Float16* p) {
  v8h lo = *(const v8h*)(p);
  v8h hi = *(const v8h*)(p + 16);
  return __builtin_shufflevector(lo, hi, 0,1,2,3,4,5,6,7,8,9,10,11,12,13,14,15);
}
__device__ __forceinline__ v8f wmma_f16(v16h a, v16h b, v8f c) {
  return __builtin_amdgcn_wmma_f32_16x16x32_f16(false, a, false, b, (short)0, c,
                                                false, false);
}

// ---------------------------------------------------------------------------
// f32 -> f16 cast
// ---------------------------------------------------------------------------
__global__ void cast_f32_f16(const float* __restrict__ src,
                             _Float16* __restrict__ dst, int n) {
  int i = blockIdx.x * blockDim.x + threadIdx.x;
  if (i < n) dst[i] = (_Float16)src[i];
}

// ---------------------------------------------------------------------------
// im2col for the stride-2 SR conv.  Patch layout k = c*4 + i*2 + j matches
// flat sr_w (C_out, C_in, 2, 2), so the conv becomes a plain GEMM.
// ---------------------------------------------------------------------------
__global__ void im2col_sr(const float* __restrict__ x, _Float16* __restrict__ P) {
  size_t idx = (size_t)blockIdx.x * blockDim.x + threadIdx.x;
  const size_t total = (size_t)B_ * M_ * 1024;
  if (idx >= total) return;
  int k = (int)(idx & 1023);
  size_t r = idx >> 10;
  int m = (int)(r % M_);
  int b = (int)(r / M_);
  int c = k >> 2, i = (k >> 1) & 1, j = k & 1;
  int tok;
  if (m < NTKV_) {                       // template: 16x16 -> 8x8
    int oh = m >> 3, ow = m & 7;
    tok = (oh * 2 + i) * 16 + (ow * 2 + j);
  } else {                               // search: 32x32 -> 16x16
    int ms = m - NTKV_;
    int oh = ms >> 4, ow = ms & 15;
    tok = NTQ_ + (oh * 2 + i) * 32 + (ow * 2 + j);
  }
  P[idx] = (_Float16)x[((size_t)b * N_ + tok) * C_ + c];
}

// ---------------------------------------------------------------------------
// Generic WMMA GEMM:  out[m,n] = scale * sum_k A[m,k]*W[n,k]  (+ bias[n]).
// Block = 128 threads = 4 waves; each wave computes a 32(m) x 64(n) tile
// (two A fragments share four B fragments -> 8 WMMAs per k-step, halving
// B traffic per FLOP).  All six fragment loads of a k-step are issued as one
// batch before the WMMA burst so the scheduler can use partial LOADcnt waits
// and issue WMMAs back-to-back (ISA 7.5 XDL issue).
// ---------------------------------------------------------------------------
template <bool OUT_F16>
__global__ __launch_bounds__(128) void gemm_wmma(
    const _Float16* __restrict__ A, const _Float16* __restrict__ W,
    void* __restrict__ outp, const float* __restrict__ bias, float scale,
    int K, int ldn) {
  const int lane = threadIdx.x & 31;
  const int wid  = threadIdx.x >> 5;
  const int hl   = lane >> 4;
  const int l16  = lane & 15;
  const int m0 = (blockIdx.x * 4 + wid) * 32;
  const int n0 = blockIdx.y * 64;

  v8f acc[2][4] = {};
  for (int k0 = 0; k0 < K; k0 += 32) {
    const _Float16* ap = A + (size_t)(m0 + l16) * K + k0 + hl * 8;
    const _Float16* wp = W + (size_t)(n0 + l16) * K + k0 + hl * 8;
    // batch all loads of this k-step
    v16h a0 = load_fragA(ap);
    v16h a1 = load_fragA(ap + (size_t)16 * K);
    v16h b0 = load_fragB(wp);
    v16h b1 = load_fragB(wp + (size_t)16 * K);
    v16h b2 = load_fragB(wp + (size_t)32 * K);
    v16h b3 = load_fragB(wp + (size_t)48 * K);
    if (k0 + 32 < K) {  // hint next A k-block into cache (global_prefetch_b8)
      __builtin_prefetch(ap + 32, 0, 1);
      __builtin_prefetch(ap + (size_t)16 * K + 32, 0, 1);
    }
    acc[0][0] = wmma_f16(a0, b0, acc[0][0]);
    acc[1][0] = wmma_f16(a1, b0, acc[1][0]);
    acc[0][1] = wmma_f16(a0, b1, acc[0][1]);
    acc[1][1] = wmma_f16(a1, b1, acc[1][1]);
    acc[0][2] = wmma_f16(a0, b2, acc[0][2]);
    acc[1][2] = wmma_f16(a1, b2, acc[1][2]);
    acc[0][3] = wmma_f16(a0, b3, acc[0][3]);
    acc[1][3] = wmma_f16(a1, b3, acc[1][3]);
  }
#pragma unroll
  for (int u = 0; u < 2; ++u) {
#pragma unroll
    for (int t = 0; t < 4; ++t) {
#pragma unroll
      for (int r = 0; r < 8; ++r) {
        int row = m0 + u * 16 + r + hl * 8;
        int col = n0 + t * 16 + l16;
        float v = acc[u][t][r] * scale;
        if (bias) v += bias[col];
        if (OUT_F16)
          ((_Float16*)outp)[(size_t)row * ldn + col] = (_Float16)v;
        else
          ((float*)outp)[(size_t)row * ldn + col] = v;
      }
    }
  }
}

// ---------------------------------------------------------------------------
// Fused SR-conv GEMM + bias + LayerNorm + f16 cast.
// Block = 128 threads = 4 waves; waves split the N dimension (wave w covers
// cols w*64..w*64+63) so one block owns 16 complete rows -> LN can be fused
// in-LDS with no f32 round-trip through HBM.
//   A: (B*M, 1024) im2col patches, W: (256, 1024) flat sr_w.
// ---------------------------------------------------------------------------
__global__ __launch_bounds__(128) void gemm_sr_ln(
    const _Float16* __restrict__ A, const _Float16* __restrict__ W,
    const float* __restrict__ bias, const float* __restrict__ g,
    const float* __restrict__ bt, _Float16* __restrict__ out) {
  __shared__ float T[16 * C_];       // 16 rows x 256 features
  __shared__ float part[2 * 16 * 8]; // partial sums / sumsq
  const int lane = threadIdx.x & 31;
  const int wid  = threadIdx.x >> 5;
  const int hl   = lane >> 4;
  const int l16  = lane & 15;
  const int m0 = blockIdx.x * 16;
  const int n0 = wid * 64;

  v8f acc[4] = {};
  for (int k0 = 0; k0 < 1024; k0 += 32) {
    const _Float16* ap = A + (size_t)(m0 + l16) * 1024 + k0 + hl * 8;
    const _Float16* wp = W + (size_t)(n0 + l16) * 1024 + k0 + hl * 8;
    v16h a  = load_fragA(ap);
    v16h b0 = load_fragB(wp);
    v16h b1 = load_fragB(wp + (size_t)16 * 1024);
    v16h b2 = load_fragB(wp + (size_t)32 * 1024);
    v16h b3 = load_fragB(wp + (size_t)48 * 1024);
    if (k0 + 32 < 1024) __builtin_prefetch(ap + 32, 0, 1);
    acc[0] = wmma_f16(a, b0, acc[0]);
    acc[1] = wmma_f16(a, b1, acc[1]);
    acc[2] = wmma_f16(a, b2, acc[2]);
    acc[3] = wmma_f16(a, b3, acc[3]);
  }
#pragma unroll
  for (int t = 0; t < 4; ++t) {
#pragma unroll
    for (int r = 0; r < 8; ++r) {
      int col = n0 + t * 16 + l16;
      T[(r + hl * 8) * C_ + col] = acc[t][r] + bias[col];
    }
  }
  __syncthreads();
  // LN: 8 threads per row, 32 features each
  const int row = threadIdx.x >> 3;
  const int grp = threadIdx.x & 7;
  {
    float s = 0.f, ss = 0.f;
    for (int f = grp * 32; f < grp * 32 + 32; ++f) {
      float v = T[row * C_ + f];
      s += v; ss += v * v;
    }
    part[row * 8 + grp] = s;
    part[128 + row * 8 + grp] = ss;
  }
  __syncthreads();
  {
    float s = 0.f, ss = 0.f;
#pragma unroll
    for (int q = 0; q < 8; ++q) {
      s += part[row * 8 + q];
      ss += part[128 + row * 8 + q];
    }
    float mean = s * (1.f / C_);
    float var  = ss * (1.f / C_) - mean * mean;
    float rstd = rsqrtf(var + LN_EPS);
    for (int f = grp * 32; f < grp * 32 + 32; ++f) {
      float v = (T[row * C_ + f] - mean) * rstd * g[f] + bt[f];
      out[(size_t)(m0 + row) * C_ + f] = (_Float16)v;
    }
  }
}

// ---------------------------------------------------------------------------
// V transpose: vt[b,h,d,m] = kv[b,m, 256 + h*32 + d]
// ---------------------------------------------------------------------------
__global__ void transpose_v(const _Float16* __restrict__ kv,
                            _Float16* __restrict__ vt) {
  size_t idx = (size_t)blockIdx.x * blockDim.x + threadIdx.x;
  const size_t total = (size_t)B_ * H_ * HD_ * M_;
  if (idx >= total) return;
  int m = (int)(idx % M_);
  size_t r = idx / M_;
  int d = (int)(r % HD_); r /= HD_;
  int h = (int)(r % H_);
  int b = (int)(r / H_);
  vt[idx] = kv[((size_t)b * M_ + m) * (2 * C_) + C_ + h * HD_ + d];
}

// ---------------------------------------------------------------------------
// Attention: one wave (32 threads) per (b, h, 16-query tile).
// tiles 0..15  -> template queries (rows tile*16),     M = 64 keys
// tiles 16..79 -> search queries (rows 256+(t-16)*16), M = 320 keys
// Phase A: S = q*k^T via WMMA into LDS.  Phase B: row softmax with both
// lane-halves scanning half a row each (cross-half combine via shfl_xor).
// Phase C: O = P*V via WMMA on pre-transposed V, scaled by 1/rowsum.
// qh already carries the 1/sqrt(hd) scale.
// ---------------------------------------------------------------------------
__global__ __launch_bounds__(32) void attention_wmma(
    const _Float16* __restrict__ qh, const _Float16* __restrict__ kv,
    const _Float16* __restrict__ vt, _Float16* __restrict__ oh) {
  __shared__ float S[16 * M_];
  __shared__ float rs[16];
  const int bid  = blockIdx.x;
  const int tile = bid % 80;
  const int bh   = bid / 80;
  const int h    = bh % H_;
  const int b    = bh / H_;
  int qrow0, M;
  if (tile < 16) { qrow0 = tile * 16;               M = NTKV_; }
  else           { qrow0 = NTQ_ + (tile - 16) * 16; M = M_;    }

  const int lane = threadIdx.x & 31;
  const int hl   = lane >> 4;
  const int l16  = lane & 15;

  // A operand: q tile (16 x 32); K dim == head dim, single WMMA per key tile
  v16h aq = load_fragA(qh + ((size_t)b * N_ + qrow0 + l16) * C_ + h * HD_ + hl * 8);

  for (int kt = 0; kt < M / 16; ++kt) {
    v16h bk = load_fragB(kv + ((size_t)b * M_ + kt * 16 + l16) * (2 * C_) +
                         h * HD_ + hl * 8);
    v8f c = v8f{};
    c = wmma_f16(aq, bk, c);
#pragma unroll
    for (int r = 0; r < 8; ++r)
      S[(r + hl * 8) * M_ + kt * 16 + l16] = c[r];
  }
  __syncthreads();

  // softmax: lane handles row l16, columns [hl*M/2, hl*M/2 + M/2)
  {
    const int row = l16;
    const int jb  = hl * (M >> 1);
    float mx = -1e30f;
    for (int j = 0; j < (M >> 1); ++j)
      mx = fmaxf(mx, S[row * M_ + jb + j]);
    mx = fmaxf(mx, __shfl_xor(mx, 16, 32));   // combine halves
    float sum = 0.f;
    for (int j = 0; j < (M >> 1); ++j) {
      float e = __expf(S[row * M_ + jb + j] - mx);
      S[row * M_ + jb + j] = e;
      sum += e;
    }
    sum += __shfl_xor(sum, 16, 32);
    if (lane < 16) rs[row] = 1.f / sum;
  }
  __syncthreads();

  v8f acc0 = v8f{}, acc1 = v8f{};
  for (int mt = 0; mt < M / 32; ++mt) {
    v16h pa;
    const int kb = mt * 32 + hl * 8;
#pragma unroll
    for (int e = 0; e < 8; ++e) pa[e] = (_Float16)S[l16 * M_ + kb + e];
#pragma unroll
    for (int e = 0; e < 8; ++e) pa[8 + e] = (_Float16)S[l16 * M_ + kb + 16 + e];
    const _Float16* vb =
        vt + ((size_t)(b * H_ + h) * HD_) * M_ + mt * 32 + hl * 8;
    v16h b0 = load_fragB(vb + (size_t)l16 * M_);
    v16h b1 = load_fragB(vb + (size_t)(16 + l16) * M_);
    acc0 = wmma_f16(pa, b0, acc0);
    acc1 = wmma_f16(pa, b1, acc1);
  }
#pragma unroll
  for (int r = 0; r < 8; ++r) {
    const int row = r + hl * 8;
    const float rcp = rs[row];
    size_t o = ((size_t)b * N_ + qrow0 + row) * C_ + h * HD_;
    oh[o + l16]      = (_Float16)(acc0[r] * rcp);
    oh[o + 16 + l16] = (_Float16)(acc1[r] * rcp);
  }
}

// ---------------------------------------------------------------------------
// Host-side orchestration
// ---------------------------------------------------------------------------
extern "C" void kernel_launch(void* const* d_in, const int* in_sizes, int n_in,
                              void* d_out, int out_size, void* d_ws,
                              size_t ws_size, hipStream_t stream) {
  const float* x      = (const float*)d_in[0];   // (B, N, C)
  const float* Wq     = (const float*)d_in[1];   // (C, C)
  const float* Wkv    = (const float*)d_in[2];   // (2C, C)
  const float* sr_w   = (const float*)d_in[3];   // (C, C, 2, 2)
  const float* sr_b   = (const float*)d_in[4];   // (C,)
  const float* ln_g   = (const float*)d_in[5];   // (C,)
  const float* ln_b   = (const float*)d_in[6];   // (C,)
  const float* proj_w = (const float*)d_in[7];   // (C, C)
  const float* proj_b = (const float*)d_in[8];   // (C,)
  float* out = (float*)d_out;                    // (B, N, C) f32

  char* ws = (char*)d_ws;
  size_t off = 0;
  auto alloc = [&](size_t bytes) {
    void* p = ws + off;
    off += (bytes + 255) & ~(size_t)255;
    return p;
  };

  const size_t nBN = (size_t)B_ * N_;   // 40960 rows
  const size_t nBM = (size_t)B_ * M_;   // 10240 rows
  _Float16* xh   = (_Float16*)alloc(nBN * C_ * 2);        // x in f16
  _Float16* qh   = (_Float16*)alloc(nBN * C_ * 2);        // scaled q, f16
  _Float16* Ph   = (_Float16*)alloc(nBM * 1024 * 2);      // im2col patches
  _Float16* xrh  = (_Float16*)alloc(nBM * C_ * 2);        // post-LN, f16
  _Float16* kvh  = (_Float16*)alloc(nBM * 2 * C_ * 2);    // K|V per token
  _Float16* vt   = (_Float16*)alloc((size_t)B_ * H_ * HD_ * M_ * 2);
  _Float16* wqh  = (_Float16*)alloc((size_t)C_ * C_ * 2);
  _Float16* wkvh = (_Float16*)alloc((size_t)2 * C_ * C_ * 2);
  _Float16* wsrh = (_Float16*)alloc((size_t)C_ * 1024 * 2);
  _Float16* wpjh = (_Float16*)alloc((size_t)C_ * C_ * 2);
  // alias (lifetimes disjoint under stream ordering): attention output
  // reuses the im2col buffer, which is consumed by gemm_sr_ln beforehand.
  _Float16* oh = Ph;

  // 1) casts
  {
    int n = (int)(nBN * C_);
    cast_f32_f16<<<(n + 255) / 256, 256, 0, stream>>>(x, xh, n);
    cast_f32_f16<<<(C_ * C_ + 255) / 256, 256, 0, stream>>>(Wq, wqh, C_ * C_);
    cast_f32_f16<<<(2 * C_ * C_ + 255) / 256, 256, 0, stream>>>(Wkv, wkvh,
                                                                2 * C_ * C_);
    cast_f32_f16<<<(C_ * 1024 + 255) / 256, 256, 0, stream>>>(sr_w, wsrh,
                                                              C_ * 1024);
    cast_f32_f16<<<(C_ * C_ + 255) / 256, 256, 0, stream>>>(proj_w, wpjh,
                                                            C_ * C_);
  }
  // 2) im2col for SR conv
  {
    size_t n = (size_t)B_ * M_ * 1024;
    im2col_sr<<<(unsigned)((n + 255) / 256), 256, 0, stream>>>(x, Ph);
  }
  // 3) Q projection (fold 1/sqrt(hd) into q), f16 out
  gemm_wmma<true><<<dim3((unsigned)(nBN / 128), C_ / 64), 128, 0, stream>>>(
      xh, wqh, qh, nullptr, QSCALE, C_, C_);
  // 4) fused SR conv GEMM + bias + LayerNorm -> xr f16 (no f32 round-trip)
  gemm_sr_ln<<<(unsigned)(nBM / 16), 128, 0, stream>>>(Ph, wsrh, sr_b, ln_g,
                                                       ln_b, xrh);
  // 5) KV projection, f16 out (N = 512 cols)
  gemm_wmma<true><<<dim3((unsigned)(nBM / 128), (2 * C_) / 64), 128, 0,
                    stream>>>(xrh, wkvh, kvh, nullptr, 1.0f, C_, 2 * C_);
  // 6) transpose V for P*V operand layout
  {
    size_t n = (size_t)B_ * H_ * HD_ * M_;
    transpose_v<<<(unsigned)((n + 255) / 256), 256, 0, stream>>>(kvh, vt);
  }
  // 7) attention (template + search tiles), f16 out into oh
  attention_wmma<<<(unsigned)(B_ * H_ * 80), 32, 0, stream>>>(qh, kvh, vt, oh);
  // 8) output projection (+ bias), f32 out
  gemm_wmma<false><<<dim3((unsigned)(nBN / 128), C_ / 64), 128, 0, stream>>>(
      oh, wpjh, out, proj_b, 1.0f, C_, C_);
}
